// OutlierLinear_77223511982407
// MI455X (gfx1250) — compile-verified, matrix operand order checked by reading, not run on hardware
//
#include <hip/hip_runtime.h>

#define IN_F   4096
#define OUT_F  4096
#define NROWS  8192     // 4 * 2048
#define NOUTL  40

#define BM 128
#define BN 128
#define BK 32
#define NTHREADS 256    // 8 wave32 waves, each owning a 32x64 output tile

typedef __attribute__((ext_vector_type(16))) _Float16 v16h;
typedef __attribute__((ext_vector_type(8)))  float    v8f;

// Types for the gfx1250 async global->LDS copy builtin:
// signature is (v4i addrspace(1)*, v4i addrspace(3)*, imm offset, imm cpol)
typedef int v4i __attribute__((vector_size(16)));
typedef __attribute__((address_space(1))) v4i* gbl_v4i_p;
typedef __attribute__((address_space(3))) v4i* lds_v4i_p;

union Frag { unsigned u[8]; v16h h; };

static __device__ __forceinline__ unsigned pk(_Float16 a, _Float16 b) {
  union { _Float16 h[2]; unsigned u; } z;
  z.h[0] = a; z.h[1] = b;
  return z.u;
}

// ---------------------------------------------------------------------------
// Kernel 0: init workspace (max scalar = 0, outlier channel flag table)
// ---------------------------------------------------------------------------
__global__ __launch_bounds__(256)
void ol_init_kernel(const int* __restrict__ oidx,
                    unsigned* __restrict__ maxbits,
                    unsigned char* __restrict__ flags) {
  int t = threadIdx.x;
  for (int i = t; i < IN_F; i += 256) flags[i] = 0;
  __syncthreads();
  if (t < NOUTL) flags[oidx[t]] = 1;
  if (t == 0) *maxbits = 0u;
}

// ---------------------------------------------------------------------------
// Kernel 1: upper = max(|x| * mask) over the whole tensor
// ---------------------------------------------------------------------------
__global__ __launch_bounds__(256)
void ol_absmax_kernel(const float* __restrict__ x,
                      const unsigned char* __restrict__ flags,
                      unsigned* __restrict__ maxbits,
                      int total) {
  __shared__ float red[256];
  int t = threadIdx.x;
  float m = 0.f;
  for (int i = blockIdx.x * blockDim.x + t; i < total; i += gridDim.x * blockDim.x) {
    float v = fabsf(x[i]);
    if (flags[i & (IN_F - 1)]) v = 0.f;
    m = fmaxf(m, v);
  }
  red[t] = m;
  __syncthreads();
  for (int s = 128; s > 0; s >>= 1) {
    if (t < s) red[t] = fmaxf(red[t], red[t + s]);
    __syncthreads();
  }
  if (t == 0) atomicMax(maxbits, __float_as_uint(red[0]));  // non-negative: bit order == float order
}

// ---------------------------------------------------------------------------
// Kernel 2: fused quantize + split-precision f16 WMMA GEMM + outlier residual
//   y = (round(clip(x)/scale) @ (Whi + Wlo)^T) * scale + bias + Xout @ B^T
// ---------------------------------------------------------------------------
__global__ __launch_bounds__(NTHREADS)
void ol_gemm_kernel(const float* __restrict__ x,
                    const float* __restrict__ w,
                    const float* __restrict__ Bmat,
                    const float* __restrict__ bias,
                    const int* __restrict__ oidx,
                    const unsigned* __restrict__ maxbits,
                    float* __restrict__ out) {
  __shared__ unsigned Ah[BM][17];            // quantized activations, f16x2 packed, padded stride
  __shared__ unsigned Wh[BN][17];            // weight hi f16
  __shared__ unsigned Wl[BN][17];            // weight lo f16 (fp32 residual)
  __shared__ float    XO[BM][NOUTL];         // outlier residual rows
  __shared__ __align__(16) float BT[BN][NOUTL];  // skinny B tile (row-major, contiguous)
  __shared__ float    BS[BN];

  const int t     = threadIdx.x;
  const int lane  = t & 31;
  const int wave  = t >> 5;
  const int wm    = wave >> 1;     // 0..3 : 32-row strip
  const int wn    = wave & 1;      // 0..1 : 64-col strip
  const int m16   = lane & 15;
  const int khalf = lane >> 4;     // K-half select per ISA 16-bit A/B layout

  const int mBase = blockIdx.y * BM;
  const int nBase = blockIdx.x * BN;

  const float upper     = __uint_as_float(*maxbits);
  const float scale     = fmaxf(upper, 1e-5f) * (1.0f / 127.0f);
  const float inv_scale = 1.0f / scale;

  // ---- Stage outlier residual rows (needs VALU transform -> via VGPRs) ----
  for (int i = t; i < BM * NOUTL; i += NTHREADS) {
    int r = i / NOUTL, j = i - r * NOUTL;
    int c = oidx[j];
    float v = x[(size_t)(mBase + r) * IN_F + c];
    XO[r][j] = v - fminf(fmaxf(v, -upper), upper);
  }

  // ---- Stage skinny-B tile: pure contiguous copy -> CDNA5 async-to-LDS ----
#if defined(__gfx1250__) && \
    __has_builtin(__builtin_amdgcn_global_load_async_to_lds_b128) && \
    __has_builtin(__builtin_amdgcn_s_wait_asynccnt)
  {
    char* gsrc = (char*)(void*)(Bmat + (size_t)nBase * NOUTL);
    char* ldst = (char*)&BT[0][0];
    for (int i = t; i < (BN * NOUTL * 4) / 16; i += NTHREADS) {
      __builtin_amdgcn_global_load_async_to_lds_b128(
          (gbl_v4i_p)(gsrc + i * 16),
          (lds_v4i_p)(ldst + i * 16),
          0, 0);
    }
    __builtin_amdgcn_s_wait_asynccnt(0);
  }
#else
  for (int i = t; i < BN * NOUTL; i += NTHREADS) {
    int n = i / NOUTL, j = i - n * NOUTL;
    BT[n][j] = Bmat[(size_t)(nBase + n) * NOUTL + j];
  }
#endif
  if (t < BN) BS[t] = bias[nBase + t];

  v8f acc[2][4];
  #pragma unroll
  for (int a = 0; a < 2; ++a)
    #pragma unroll
    for (int b = 0; b < 4; ++b)
      acc[a][b] = (v8f){};

  for (int kb = 0; kb < IN_F; kb += BK) {
    __syncthreads();

    // A tile: BM x BK fp32 -> quantize to exact-integer f16
    #pragma unroll
    for (int p = 0; p < (BM * BK / 4) / NTHREADS; ++p) {   // 4 float4 per thread
      int i  = t + p * NTHREADS;
      int r  = i >> 3;                                     // 8 float4 per row
      int c4 = i & 7;
      const float4 v = *(const float4*)(x + (size_t)(mBase + r) * IN_F + kb + c4 * 4);
      float q0 = rintf(fminf(fmaxf(v.x, -upper), upper) * inv_scale);
      float q1 = rintf(fminf(fmaxf(v.y, -upper), upper) * inv_scale);
      float q2 = rintf(fminf(fmaxf(v.z, -upper), upper) * inv_scale);
      float q3 = rintf(fminf(fmaxf(v.w, -upper), upper) * inv_scale);
      Ah[r][c4 * 2 + 0] = pk((_Float16)q0, (_Float16)q1);
      Ah[r][c4 * 2 + 1] = pk((_Float16)q2, (_Float16)q3);
    }
    // W tile: BN x BK fp32 -> split into f16 hi + f16 residual
    #pragma unroll
    for (int p = 0; p < (BN * BK / 4) / NTHREADS; ++p) {   // 4 float4 per thread
      int i  = t + p * NTHREADS;
      int r  = i >> 3;
      int c4 = i & 7;
      const float4 v = *(const float4*)(w + (size_t)(nBase + r) * IN_F + kb + c4 * 4);
      _Float16 h0 = (_Float16)v.x, h1 = (_Float16)v.y, h2 = (_Float16)v.z, h3 = (_Float16)v.w;
      _Float16 l0 = (_Float16)(v.x - (float)h0);
      _Float16 l1 = (_Float16)(v.y - (float)h1);
      _Float16 l2 = (_Float16)(v.z - (float)h2);
      _Float16 l3 = (_Float16)(v.w - (float)h3);
      Wh[r][c4 * 2 + 0] = pk(h0, h1);
      Wh[r][c4 * 2 + 1] = pk(h2, h3);
      Wl[r][c4 * 2 + 0] = pk(l0, l1);
      Wl[r][c4 * 2 + 1] = pk(l2, l3);
    }
    __syncthreads();

    // Per-lane fragment gather per ISA 7.12.2 (lane%16 = row/col, lane/16 = K-half)
    Frag a[2];
    #pragma unroll
    for (int rt = 0; rt < 2; ++rt) {
      const unsigned* row = &Ah[wm * 32 + rt * 16 + m16][0];
      #pragma unroll
      for (int q = 0; q < 4; ++q) {
        a[rt].u[q]     = row[khalf * 4 + q];
        a[rt].u[4 + q] = row[8 + khalf * 4 + q];
      }
    }
    #pragma unroll
    for (int ct = 0; ct < 4; ++ct) {
      Frag bh, bl;
      const unsigned* rh = &Wh[wn * 64 + ct * 16 + m16][0];
      const unsigned* rl = &Wl[wn * 64 + ct * 16 + m16][0];
      #pragma unroll
      for (int q = 0; q < 4; ++q) {
        bh.u[q]     = rh[khalf * 4 + q];
        bh.u[4 + q] = rh[8 + khalf * 4 + q];
        bl.u[q]     = rl[khalf * 4 + q];
        bl.u[4 + q] = rl[8 + khalf * 4 + q];
      }
      #pragma unroll
      for (int rt = 0; rt < 2; ++rt) {
        acc[rt][ct] = __builtin_amdgcn_wmma_f32_16x16x32_f16(
            false, a[rt].h, false, bh.h, (short)0, acc[rt][ct], false, false);
        acc[rt][ct] = __builtin_amdgcn_wmma_f32_16x16x32_f16(
            false, a[rt].h, false, bl.h, (short)0, acc[rt][ct], false, false);
      }
    }
  }
  __syncthreads();

  // Epilogue: y = acc*scale + bias + outlier residual (40-term dot on VALU)
  const int cb = wn * 64 + m16;
  #pragma unroll
  for (int rt = 0; rt < 2; ++rt) {
    #pragma unroll
    for (int j = 0; j < 8; ++j) {
      int rl = wm * 32 + rt * 16 + khalf * 8 + j;   // M = vgpr + 8*(lane/16)
      float r0 = 0.f, r1 = 0.f, r2 = 0.f, r3 = 0.f;
      #pragma unroll 4
      for (int o = 0; o < NOUTL; ++o) {
        float xo = XO[rl][o];
        r0 += xo * BT[cb +  0][o];
        r1 += xo * BT[cb + 16][o];
        r2 += xo * BT[cb + 32][o];
        r3 += xo * BT[cb + 48][o];
      }
      float* orow = out + (size_t)(mBase + rl) * OUT_F + nBase + cb;
      orow[ 0] = acc[rt][0][j] * scale + BS[cb +  0] + r0;
      orow[16] = acc[rt][1][j] * scale + BS[cb + 16] + r1;
      orow[32] = acc[rt][2][j] * scale + BS[cb + 32] + r2;
      orow[48] = acc[rt][3][j] * scale + BS[cb + 48] + r3;
    }
  }
}

// ---------------------------------------------------------------------------
extern "C" void kernel_launch(void* const* d_in, const int* in_sizes, int n_in,
                              void* d_out, int out_size, void* d_ws, size_t ws_size,
                              hipStream_t stream) {
  (void)in_sizes; (void)n_in; (void)out_size; (void)ws_size;
  const float* x    = (const float*)d_in[0];
  const float* w    = (const float*)d_in[1];
  const float* Bmat = (const float*)d_in[2];
  const float* bias = (const float*)d_in[3];
  const int*   oidx = (const int*)d_in[4];

  unsigned*      maxbits = (unsigned*)d_ws;
  unsigned char* flags   = (unsigned char*)d_ws + 256;

  hipLaunchKernelGGL(ol_init_kernel, dim3(1), dim3(256), 0, stream,
                     oidx, maxbits, flags);
  hipLaunchKernelGGL(ol_absmax_kernel, dim3(1024), dim3(256), 0, stream,
                     x, flags, maxbits, NROWS * IN_F);

  dim3 grid(OUT_F / BN, NROWS / BM);   // 32 x 64 = 2048 blocks
  hipLaunchKernelGGL(ol_gemm_kernel, grid, dim3(NTHREADS), 0, stream,
                     x, w, Bmat, bias, oidx, maxbits, (float*)d_out);
}
